// Decoder_30580167148063
// MI455X (gfx1250) — matrix operand order repaired and checked by reading, not compiled, not run on hardware
//
#include <hip/hip_runtime.h>
#include <hip/hip_bf16.h>
#include <math.h>

typedef __bf16 bf16_t;
typedef __attribute__((ext_vector_type(16))) __bf16 v16bf;
typedef __attribute__((ext_vector_type(8)))  __bf16 v8bf;
typedef __attribute__((ext_vector_type(8)))  float  v8f;

#define B_     128
#define TENC   512
#define INFEAT 256
#define TDEC   200
#define PRE1N  256
#define PRE2N  128
#define HIDN   256
#define OUTD   400

// ---------------- device helpers ----------------

__device__ inline float fast_tanh(float x) {
#if __has_builtin(__builtin_amdgcn_tanhf)
  return __builtin_amdgcn_tanhf(x);     // v_tanh_f32 on gfx1250
#else
  return tanhf(x);
#endif
}
__device__ inline float sigmoid_(float x) { return 1.0f / (1.0f + __expf(-x)); }

// A-fragment (16x32 bf16), LDS row already bf16 (row M = lane%16).
// lanes 0-15:  e<8 -> K=k0+e,    e>=8 -> K=k0+16+(e-8)
// lanes 16-31: e<8 -> K=k0+8+e,  e>=8 -> K=k0+24+(e-8)
__device__ inline v16bf load_a_frag(const bf16_t* __restrict__ row, int k0, int half) {
  v8bf a0 = *(const v8bf*)(row + k0 + 8 * half);        // 16B-aligned ds_load_b128
  v8bf a1 = *(const v8bf*)(row + k0 + 16 + 8 * half);   // 16B-aligned ds_load_b128
  v16bf a;
#pragma unroll
  for (int i = 0; i < 8; ++i) { a[i] = a0[i]; a[8 + i] = a1[i]; }
  return a;
}

// B-fragment: pre-tiled weights, 32 lanes x 16 bf16 contiguous per 16(N)x32(K) tile.
__device__ inline v16bf load_b_frag(const bf16_t* __restrict__ Wt, int nt, int kt, int KT, int lane) {
  return *(const v16bf*)(Wt + (((long)nt * KT + kt) * 32 + lane) * 16);
}

__device__ inline v8f wmma_bf16(v16bf a, v16bf b, v8f c) {
  return __builtin_amdgcn_wmma_f32_16x16x32_bf16(false, a, false, b, (short)0, c, false, false);
}

// Stage 16 rows of activations into LDS as bf16, fusing an optional feature-concat
// [A1 (n1 cols) | A2 (n2 cols) | zero-pad to Kp].  Packed pair stores (ds_store_b32).
__device__ inline void stage_rows(bf16_t* dst,
                                  const float* srcA, long lda, int na,
                                  const float* srcB, long ldb, int nb,
                                  int mtile, int Kp) {
  int tot = 16 * Kp;
  for (int i = 2 * (int)threadIdx.x; i < tot; i += 2 * (int)blockDim.x) {
    int r = i / Kp, k = i - r * Kp;       // Kp even -> pair stays in one row
    long gr = mtile * 16 + r;
    float v0 = 0.0f, v1 = 0.0f;
    if (srcA && k < na)                    v0 = srcA[gr * lda + k];
    else if (srcB && k >= na && k < na+nb) v0 = srcB[gr * ldb + (k - na)];
    int k1 = k + 1;
    if (srcA && k1 < na)                    v1 = srcA[gr * lda + k1];
    else if (srcB && k1 >= na && k1 < na+nb) v1 = srcB[gr * ldb + (k1 - na)];
    union { struct { __bf16 lo, hi; } h; unsigned u; } pk;
    pk.h.lo = (__bf16)v0; pk.h.hi = (__bf16)v1;
    *(unsigned*)(dst + i) = pk.u;
  }
}

// ---------------- weight pre-tiling (fp32 [N,K] row-major -> bf16 WMMA tiles) ----------------

__global__ void tile_weights_kernel(const float* __restrict__ W, bf16_t* __restrict__ Wt,
                                    int N, int K, int KT) {
  int nt = blockIdx.x, kt = blockIdx.y;
  int lane = threadIdx.x >> 4;          // 0..31
  int e    = threadIdx.x & 15;          // 0..15
  int n = nt * 16 + (lane & 15);
  int k = kt * 32 + ((lane >= 16) ? 8 : 0) + (e & 7) + ((e >= 8) ? 16 : 0);
  float v = (n < N && k < K) ? W[(long)n * K + k] : 0.0f;
  Wt[(((long)nt * KT + kt) * 32 + lane) * 16 + e] = (bf16_t)v;
}

// ---------------- generic WMMA GEMM: C = act([A1|A2] @ W^T + bias) ----------------
// grid = (Mtiles, Ntiles/WPB), block = 32*WPB, dyn LDS = 16*Kp*2 bytes.

template <int ACT, typename OUT>
__global__ void gemm_kernel(const float* __restrict__ A1, long lda1, int n1,
                            const float* __restrict__ A2, long lda2, int n2, int Kp,
                            const bf16_t* __restrict__ Wt, const float* __restrict__ bias,
                            OUT* __restrict__ C, long ldc) {
  extern __shared__ char smem[];
  bf16_t* lds = (bf16_t*)smem;
  int mt = blockIdx.x;
  stage_rows(lds, A1, lda1, n1, A2, lda2, n2, mt, Kp);
  __syncthreads();
  int wave = threadIdx.x >> 5, lane = threadIdx.x & 31;
  int WPB = blockDim.x >> 5;
  int nt = blockIdx.y * WPB + wave;
  int KT = Kp >> 5, half = lane >> 4, nl = lane & 15;
  const bf16_t* row = lds + nl * Kp;
  v8f acc = {};
  for (int kt = 0; kt < KT; ++kt) {
    v16bf a = load_a_frag(row, kt * 32, half);
    v16bf b = load_b_frag(Wt, nt, kt, KT, lane);
    acc = wmma_bf16(a, b, acc);
  }
  int n = nt * 16 + nl;
  float bv = bias ? bias[n] : 0.0f;
#pragma unroll
  for (int r = 0; r < 8; ++r) {
    int m = mt * 16 + r + 8 * half;
    float v = acc[r] + bv;
    if (ACT == 1) v = fmaxf(v, 0.0f);
    C[(long)m * ldc + n] = (OUT)v;
  }
}

// ---------------- fused GRU cell (torch gate order r,z,n) ----------------
// Each wave owns one 16-wide hidden tile j: computes gi/gh columns {j, j+NH, j+2NH},
// then gates + fp32 state update + optional residual (resOut = h_new + resAdd).
// X = [X1 (n1) | X2 (n2)] fused concat.  h_prev / resAdd read fp32 from global.

__global__ void gru_kernel(const float* __restrict__ X1, long ldx1, int n1,
                           const float* __restrict__ X2, long ldx2, int n2, int Kxp,
                           const bf16_t* __restrict__ Wi, const float* __restrict__ bi,
                           const float* __restrict__ Hin, const bf16_t* __restrict__ Wh,
                           const float* __restrict__ bh, float* __restrict__ Hout,
                           int NH, int NHp,
                           const float* __restrict__ resAdd, float* __restrict__ resOut) {
  extern __shared__ char smem[];
  bf16_t* ldsX = (bf16_t*)smem;
  bf16_t* ldsH = ldsX + 16 * Kxp;
  int mt = blockIdx.x;
  stage_rows(ldsX, X1, ldx1, n1, X2, ldx2, n2, mt, Kxp);
  stage_rows(ldsH, Hin, NH, NH, nullptr, 0, 0, mt, NHp);
  __syncthreads();
  int wave = threadIdx.x >> 5, lane = threadIdx.x & 31;
  int WPB = blockDim.x >> 5;
  int jt = blockIdx.y * WPB + wave;
  int NT = NH >> 4;
  int KTx = Kxp >> 5, KTh = NHp >> 5;
  int half = lane >> 4, nl = lane & 15;
  v8f ir = {}, iz = {}, in_ = {}, hr = {}, hz = {}, hn = {};
  const bf16_t* xrow = ldsX + nl * Kxp;
  for (int kt = 0; kt < KTx; ++kt) {
    v16bf a = load_a_frag(xrow, kt * 32, half);
    ir  = wmma_bf16(a, load_b_frag(Wi, jt,          kt, KTx, lane), ir);
    iz  = wmma_bf16(a, load_b_frag(Wi, jt + NT,     kt, KTx, lane), iz);
    in_ = wmma_bf16(a, load_b_frag(Wi, jt + 2 * NT, kt, KTx, lane), in_);
  }
  const bf16_t* hrow = ldsH + nl * NHp;
  for (int kt = 0; kt < KTh; ++kt) {
    v16bf a = load_a_frag(hrow, kt * 32, half);
    hr = wmma_bf16(a, load_b_frag(Wh, jt,          kt, KTh, lane), hr);
    hz = wmma_bf16(a, load_b_frag(Wh, jt + NT,     kt, KTh, lane), hz);
    hn = wmma_bf16(a, load_b_frag(Wh, jt + 2 * NT, kt, KTh, lane), hn);
  }
  int j = jt * 16 + nl;
  float bir = bi[j], biz = bi[j + NH], bin = bi[j + 2 * NH];
  float bhr = bh[j], bhz = bh[j + NH], bhn = bh[j + 2 * NH];
#pragma unroll
  for (int r = 0; r < 8; ++r) {
    int m = mt * 16 + r + 8 * half;
    float rg = sigmoid_(ir[r] + bir + hr[r] + bhr);
    float zg = sigmoid_(iz[r] + biz + hz[r] + bhz);
    float ng = fast_tanh(in_[r] + bin + rg * (hn[r] + bhn));
    float hp = Hin[(long)m * NH + j];                 // fp32-exact recurrence
    float hv = (1.0f - zg) * ng + zg * hp;
    Hout[(long)m * NH + j] = hv;
    if (resOut) resOut[(long)m * NH + j] = hv + resAdd[(long)m * NH + j];
  }
}

// ---------------- attention: scores(tanh) + softmax + context ----------------
// grid = B blocks, block = 256 threads (8 waves).

__global__ void attention_kernel(const float* __restrict__ Q, const bf16_t* __restrict__ PA,
                                 const float* __restrict__ Vw, const float* __restrict__ Inp,
                                 float* __restrict__ Ctx, float* __restrict__ AlignOut,
                                 long align_bstride) {
  __shared__ float qs[HIDN], vs[HIDN], sc[TENC], red[8], bcast[2];
  int b = blockIdx.x, tid = threadIdx.x;
  int wave = tid >> 5, lane = tid & 31;
  if (tid < HIDN) { qs[tid] = Q[(long)b * HIDN + tid]; vs[tid] = Vw[tid]; }
  __syncthreads();
  // scores[t] = sum_h v[h] * tanh(q[h] + pa[b,t,h])
  for (int t = wave; t < TENC; t += 8) {
    const bf16_t* par = PA + ((long)b * TENC + t) * HIDN;
    float p = 0.0f;
    for (int h = lane; h < HIDN; h += 32) p += vs[h] * fast_tanh(qs[h] + (float)par[h]);
    for (int o = 16; o > 0; o >>= 1) p += __shfl_xor(p, o, 32);
    if (lane == 0) sc[t] = p;
  }
  __syncthreads();
  // softmax (block max then sum)
  float m = -1e30f;
  for (int i = tid; i < TENC; i += blockDim.x) m = fmaxf(m, sc[i]);
  for (int o = 16; o > 0; o >>= 1) m = fmaxf(m, __shfl_xor(m, o, 32));
  if (lane == 0) red[wave] = m;
  __syncthreads();
  if (tid == 0) { float mm = red[0]; for (int w = 1; w < 8; ++w) mm = fmaxf(mm, red[w]); bcast[0] = mm; }
  __syncthreads();
  float mx = bcast[0], s = 0.0f;
  for (int i = tid; i < TENC; i += blockDim.x) { float e = __expf(sc[i] - mx); sc[i] = e; s += e; }
  for (int o = 16; o > 0; o >>= 1) s += __shfl_xor(s, o, 32);
  if (lane == 0) red[wave] = s;
  __syncthreads();
  if (tid == 0) { float ss = 0.0f; for (int w = 0; w < 8; ++w) ss += red[w]; bcast[1] = ss; }
  __syncthreads();
  float inv = 1.0f / bcast[1];
  for (int i = tid; i < TENC; i += blockDim.x) {
    float a = sc[i] * inv; sc[i] = a;
    AlignOut[(long)b * align_bstride + i] = a;
  }
  __syncthreads();
  // ctx[e] = sum_t align[t] * inputs[b,t,e]  (coalesced over e)
  const float* ip = Inp + (long)b * TENC * INFEAT + tid;
  float acc = 0.0f;
  for (int t = 0; t < TENC; ++t) {
    if (t + 8 < TENC) __builtin_prefetch(ip + (long)(t + 8) * INFEAT, 0, 1);
    acc += sc[t] * ip[(long)t * INFEAT];
  }
  Ctx[(long)b * INFEAT + tid] = acc;
}

// ---------------- stopnet head: sigmoid(relu(stop_h) @ s_lw^T + s_lb) ----------------

__global__ void stoplin_kernel(const float* __restrict__ SH, const float* __restrict__ lw,
                               const float* __restrict__ lb, float* __restrict__ outp, long ld) {
  int b = threadIdx.x;  // 128 threads
  float acc = lb[0];
  const float* row = SH + (long)b * OUTD;
  for (int k = 0; k < OUTD; ++k) acc += fmaxf(row[k], 0.0f) * lw[k];
  outp[(long)b * ld] = sigmoid_(acc);
}

// ---------------- host orchestration ----------------

extern "C" void kernel_launch(void* const* d_in, const int* in_sizes, int n_in,
                              void* d_out, int out_size, void* d_ws, size_t ws_size,
                              hipStream_t stream) {
  (void)in_sizes; (void)n_in; (void)out_size; (void)ws_size;
  const float* inputs = (const float*)d_in[0];
  const float* memory = (const float*)d_in[1];
  const float* p1_w = (const float*)d_in[2];  const float* p1_b = (const float*)d_in[3];
  const float* p2_w = (const float*)d_in[4];  const float* p2_b = (const float*)d_in[5];
  const float* att_wi = (const float*)d_in[6]; const float* att_bi = (const float*)d_in[7];
  const float* att_wh = (const float*)d_in[8]; const float* att_bh = (const float*)d_in[9];
  const float* q_w = (const float*)d_in[10];  const float* q_b = (const float*)d_in[11];
  const float* a_w = (const float*)d_in[12];  const float* a_b = (const float*)d_in[13];
  const float* v_w = (const float*)d_in[14];
  const float* proj_w = (const float*)d_in[15]; const float* proj_b = (const float*)d_in[16];
  const float* d1_wi = (const float*)d_in[17]; const float* d1_bi = (const float*)d_in[18];
  const float* d1_wh = (const float*)d_in[19]; const float* d1_bh = (const float*)d_in[20];
  const float* d2_wi = (const float*)d_in[21]; const float* d2_bi = (const float*)d_in[22];
  const float* d2_wh = (const float*)d_in[23]; const float* d2_bh = (const float*)d_in[24];
  const float* mel_w = (const float*)d_in[25]; const float* mel_b = (const float*)d_in[26];
  const float* s_wi = (const float*)d_in[27];  const float* s_bi = (const float*)d_in[28];
  const float* s_wh = (const float*)d_in[29];  const float* s_bh = (const float*)d_in[30];
  const float* s_lw = (const float*)d_in[31];  const float* s_lb = (const float*)d_in[32];

  float* outs   = (float*)d_out;                        // [B, 200, 400]
  float* aligns = outs + (long)B_ * TDEC * OUTD;        // [B, 200, 512]
  float* stops  = aligns + (long)B_ * TDEC * TENC;      // [B, 200, 1]

  // bump allocator over d_ws
  char* base = (char*)d_ws;
  size_t pos = 0;
  auto alloc = [&](size_t bytes) -> void* {
    void* p = base + pos;
    pos += (bytes + 255) & ~(size_t)255;
    return p;
  };
  auto allocT = [&](int N, int Kp) -> bf16_t* {
    return (bf16_t*)alloc((size_t)(N / 16) * (Kp / 32) * 512 * sizeof(bf16_t));
  };

  bf16_t* p1_t   = allocT(PRE1N, 416);
  bf16_t* p2_t   = allocT(PRE2N, 256);
  bf16_t* awi_t  = allocT(3 * HIDN, 384);
  bf16_t* awh_t  = allocT(3 * HIDN, 256);
  bf16_t* q_t    = allocT(HIDN, 256);
  bf16_t* aw_t   = allocT(INFEAT, 256);
  bf16_t* proj_t = allocT(HIDN, 512);
  bf16_t* d1wi_t = allocT(3 * HIDN, 256);
  bf16_t* d1wh_t = allocT(3 * HIDN, 256);
  bf16_t* d2wi_t = allocT(3 * HIDN, 256);
  bf16_t* d2wh_t = allocT(3 * HIDN, 256);
  bf16_t* mel_t  = allocT(OUTD, 256);
  bf16_t* swi_t  = allocT(3 * OUTD, 416);
  bf16_t* swh_t  = allocT(3 * OUTD, 416);
  bf16_t* pa     = (bf16_t*)alloc((size_t)B_ * TENC * HIDN * sizeof(bf16_t));  // proc_annot bf16

  float* pm1     = (float*)alloc((size_t)B_ * PRE1N * 4);
  float* pm2     = (float*)alloc((size_t)B_ * PRE2N * 4);
  float* qv      = (float*)alloc((size_t)B_ * HIDN * 4);
  float* ctx     = (float*)alloc((size_t)B_ * INFEAT * 4);
  float* dec_in  = (float*)alloc((size_t)B_ * HIDN * 4);
  float* x1      = (float*)alloc((size_t)B_ * HIDN * 4);
  float* dec_out = (float*)alloc((size_t)B_ * HIDN * 4);
  float* atth[2] = {(float*)alloc((size_t)B_ * HIDN * 4), (float*)alloc((size_t)B_ * HIDN * 4)};
  float* h1v[2]  = {(float*)alloc((size_t)B_ * HIDN * 4), (float*)alloc((size_t)B_ * HIDN * 4)};
  float* h2v[2]  = {(float*)alloc((size_t)B_ * HIDN * 4), (float*)alloc((size_t)B_ * HIDN * 4)};
  float* shv[2]  = {(float*)alloc((size_t)B_ * OUTD * 4), (float*)alloc((size_t)B_ * OUTD * 4)};

  auto tile = [&](const float* W, bf16_t* Wt, int N, int K, int Kp) {
    dim3 g(N / 16, Kp / 32);
    tile_weights_kernel<<<g, 512, 0, stream>>>(W, Wt, N, K, Kp / 32);
  };
  tile(p1_w, p1_t, PRE1N, 400, 416);
  tile(p2_w, p2_t, PRE2N, 256, 256);
  tile(att_wi, awi_t, 3 * HIDN, 384, 384);
  tile(att_wh, awh_t, 3 * HIDN, 256, 256);
  tile(q_w, q_t, HIDN, 256, 256);
  tile(a_w, aw_t, INFEAT, 256, 256);
  tile(proj_w, proj_t, HIDN, 512, 512);
  tile(d1_wi, d1wi_t, 3 * HIDN, 256, 256);
  tile(d1_wh, d1wh_t, 3 * HIDN, 256, 256);
  tile(d2_wi, d2wi_t, 3 * HIDN, 256, 256);
  tile(d2_wh, d2wh_t, 3 * HIDN, 256, 256);
  tile(mel_w, mel_t, OUTD, 256, 256);
  tile(s_wi, swi_t, 3 * OUTD, 400, 416);
  tile(s_wh, swh_t, 3 * OUTD, 400, 416);

  // zero initial states (re-done every call; graph-capture safe)
  (void)hipMemsetAsync(atth[0], 0, (size_t)B_ * HIDN * 4, stream);
  (void)hipMemsetAsync(h1v[0], 0, (size_t)B_ * HIDN * 4, stream);
  (void)hipMemsetAsync(h2v[0], 0, (size_t)B_ * HIDN * 4, stream);
  (void)hipMemsetAsync(ctx, 0, (size_t)B_ * INFEAT * 4, stream);
  (void)hipMemsetAsync(shv[0], 0, (size_t)B_ * OUTD * 4, stream);

  // proc_annot (bf16) = inputs @ a_w^T + a_b : M = B*TENC = 65536 rows
  gemm_kernel<0, bf16_t><<<dim3((B_ * TENC) / 16, 4), 128, 16 * 256 * 2, stream>>>(
      inputs, INFEAT, 256, nullptr, 0, 0, 256, aw_t, a_b, pa, HIDN);

  const long memLd = (long)TDEC * OUTD;  // 80000 = row stride of grouped memory & outs
  int cur = 0;
  for (int t = 0; t < TDEC; ++t) {
    const float* xin = (t == 0) ? nullptr : (memory + (long)(t - 1) * OUTD);
    // prenet (relu x2)
    gemm_kernel<1, float><<<dim3(8, 4), 128, 16 * 416 * 2, stream>>>(
        xin, memLd, 400, nullptr, 0, 0, 416, p1_t, p1_b, pm1, PRE1N);
    gemm_kernel<1, float><<<dim3(8, 2), 128, 16 * 256 * 2, stream>>>(
        pm1, PRE1N, 256, nullptr, 0, 0, 256, p2_t, p2_b, pm2, PRE2N);
    // attention GRU, input = [pm | ctx] (concat fused into staging)
    gru_kernel<<<dim3(8, 4), 128, 16 * (384 + 256) * 2, stream>>>(
        pm2, PRE2N, PRE2N, ctx, INFEAT, INFEAT, 384,
        awi_t, att_bi, atth[cur], awh_t, att_bh, atth[1 - cur], HIDN, 256,
        nullptr, nullptr);
    // query
    gemm_kernel<0, float><<<dim3(8, 4), 128, 16 * 256 * 2, stream>>>(
        atth[1 - cur], HIDN, 256, nullptr, 0, 0, 256, q_t, q_b, qv, HIDN);
    // attention: scores + softmax + ctx (writes aligns to d_out)
    attention_kernel<<<B_, 256, 0, stream>>>(
        qv, pa, v_w, inputs, ctx, aligns + (long)t * TENC, (long)TDEC * TENC);
    // decoder input = [att_h | ctx] @ proj (concat fused)
    gemm_kernel<0, float><<<dim3(8, 4), 128, 16 * 512 * 2, stream>>>(
        atth[1 - cur], HIDN, HIDN, ctx, INFEAT, INFEAT, 512,
        proj_t, proj_b, dec_in, HIDN);
    // stacked residual GRUs
    gru_kernel<<<dim3(8, 4), 128, 16 * (256 + 256) * 2, stream>>>(
        dec_in, HIDN, 256, nullptr, 0, 0, 256,
        d1wi_t, d1_bi, h1v[cur], d1wh_t, d1_bh, h1v[1 - cur], HIDN, 256,
        dec_in, x1);
    gru_kernel<<<dim3(8, 4), 128, 16 * (256 + 256) * 2, stream>>>(
        x1, HIDN, 256, nullptr, 0, 0, 256,
        d2wi_t, d2_bi, h2v[cur], d2wh_t, d2_bh, h2v[1 - cur], HIDN, 256,
        x1, dec_out);
    // mel projection straight into d_out[:, t, :]
    gemm_kernel<0, float><<<dim3(8, 5), 160, 16 * 256 * 2, stream>>>(
        dec_out, HIDN, 256, nullptr, 0, 0, 256, mel_t, mel_b,
        outs + (long)t * OUTD, memLd);
    // stopnet GRU (reads mel frame from d_out) + head
    gru_kernel<<<dim3(8, 5), 160, 16 * (416 + 416) * 2, stream>>>(
        outs + (long)t * OUTD, memLd, 400, nullptr, 0, 0, 416,
        swi_t, s_bi, shv[cur], swh_t, s_bh, shv[1 - cur], OUTD, 416,
        nullptr, nullptr);
    stoplin_kernel<<<1, 128, 0, stream>>>(shv[1 - cur], s_lw, s_lb, stops + t, (long)TDEC);
    cur ^= 1;
  }
}
